// MultiHeadAttention_69406671503932
// MI455X (gfx1250) — matrix operand order repaired and checked by reading, not compiled, not run on hardware
//
#include <hip/hip_runtime.h>
#include <hip/hip_bf16.h>

typedef __attribute__((ext_vector_type(16))) _Float16 v16h;
typedef __attribute__((ext_vector_type(8)))  _Float16 v8h;
typedef __attribute__((ext_vector_type(8)))  float    v8f;
typedef __attribute__((ext_vector_type(4)))  float    v4f;

#define B_  2
#define N_  2048
#define D_  1024
#define H_  16
#define HD_ 64

__device__ __forceinline__ v8f wmma32(v16h a, v16h b, v8f c) {
    // v_wmma_f32_16x16x32_f16: D = A(16x32 f16) * B(32x16 f16) + C(16x16 f32)
    return __builtin_amdgcn_wmma_f32_16x16x32_f16(
        /*neg_a=*/false, a, /*neg_b=*/false, b,
        /*c_mod=*/(short)0, c, /*reuse_a=*/false, /*reuse_b=*/false);
}

// ---- fragment loaders -------------------------------------------------------
// A fragment (16x32 f16) from f32 row-major matrix, row stride ld.
// Lane l<16: row m0+l, K = {k0..+7, k0+16..+23}; lane l+16: K = {+8..+15, +24..+31}.
__device__ __forceinline__ v16h load_a_f32(const float* __restrict__ base, int ld,
                                           int m0, int k0, int lane) {
    const int r = lane & 15, hi = lane >> 4;
    const float* p = base + (long)(m0 + r) * ld + k0 + hi * 8;
    v4f f0 = *(const v4f*)(p + 0);
    v4f f1 = *(const v4f*)(p + 4);
    v4f f2 = *(const v4f*)(p + 16);
    v4f f3 = *(const v4f*)(p + 20);
    v16h a;
#pragma unroll
    for (int i = 0; i < 4; ++i) {
        a[i]      = (_Float16)f0[i];
        a[4 + i]  = (_Float16)f1[i];
        a[8 + i]  = (_Float16)f2[i];
        a[12 + i] = (_Float16)f3[i];
    }
    return a;
}

// A fragment from f16 row-major matrix.
__device__ __forceinline__ v16h load_a_f16(const _Float16* __restrict__ base, int ld,
                                           int m0, int k0, int lane) {
    const int r = lane & 15, hi = lane >> 4;
    const _Float16* p = base + (long)(m0 + r) * ld + k0 + hi * 8;
    v8h x0 = *(const v8h*)(p + 0);
    v8h x1 = *(const v8h*)(p + 16);
    v16h a;
#pragma unroll
    for (int i = 0; i < 8; ++i) { a[i] = x0[i]; a[8 + i] = x1[i]; }
    return a;
}

// B fragment (32x16) from "column-major" f32 source (column n contiguous in k).
// Lane c<16: col n0+c, K = {k0..+7, k0+16..+23}; lane c+16: K = {+8..+15, +24..+31}.
__device__ __forceinline__ v16h load_b_f32(const float* __restrict__ colbase, int ldcol,
                                           int n0, int k0, int lane) {
    const int c = lane & 15, hi = lane >> 4;
    const float* p = colbase + (long)(n0 + c) * ldcol + k0 + hi * 8;
    v4f f0 = *(const v4f*)(p + 0);
    v4f f1 = *(const v4f*)(p + 4);
    v4f f2 = *(const v4f*)(p + 16);
    v4f f3 = *(const v4f*)(p + 20);
    v16h b;
#pragma unroll
    for (int i = 0; i < 4; ++i) {
        b[i]      = (_Float16)f0[i];
        b[4 + i]  = (_Float16)f1[i];
        b[8 + i]  = (_Float16)f2[i];
        b[12 + i] = (_Float16)f3[i];
    }
    return b;
}

// B fragment from column-major f16 source.
__device__ __forceinline__ v16h load_b_f16(const _Float16* __restrict__ colbase, int ldcol,
                                           int n0, int k0, int lane) {
    const int c = lane & 15, hi = lane >> 4;
    const _Float16* p = colbase + (long)(n0 + c) * ldcol + k0 + hi * 8;
    v8h x0 = *(const v8h*)(p + 0);
    v8h x1 = *(const v8h*)(p + 16);
    v16h b;
#pragma unroll
    for (int i = 0; i < 8; ++i) { b[i] = x0[i]; b[8 + i] = x1[i]; }
    return b;
}

// A fragment from a per-wave 16x32 f16 row-major LDS tile (P matrix).
__device__ __forceinline__ v16h load_a_lds(const _Float16* P, int lane) {
    const int r = lane & 15, hi = lane >> 4;
    const _Float16* p = P + r * 32 + hi * 8;
    v8h x0 = *(const v8h*)(p + 0);
    v8h x1 = *(const v8h*)(p + 16);
    v16h a;
#pragma unroll
    for (int i = 0; i < 8; ++i) { a[i] = x0[i]; a[8 + i] = x1[i]; }
    return a;
}

// ---- kernel 1: qkv = x @ w_qkv.T -> Q,K (head-major f16) and V^T -----------
// grid (M/32, 3072/64), block 32: one wave = 32x64 tile, double-buffered K loop.
__global__ __launch_bounds__(32) void qkv_kernel(const float* __restrict__ x,
                                                 const float* __restrict__ w_qkv,
                                                 _Float16* __restrict__ qb,
                                                 _Float16* __restrict__ kb,
                                                 _Float16* __restrict__ vT) {
    const int lane = threadIdx.x;
    const int m0 = blockIdx.x * 32;
    const int n0 = blockIdx.y * 64;
    v8f acc[2][4] = {{v8f{}, v8f{}, v8f{}, v8f{}}, {v8f{}, v8f{}, v8f{}, v8f{}}};

    // prologue: fragments for k0 = 0
    v16h a0 = load_a_f32(x, D_, m0, 0, lane);
    v16h a1 = load_a_f32(x, D_, m0 + 16, 0, lane);
    v16h b0 = load_b_f32(w_qkv, D_, n0 + 0,  0, lane);
    v16h b1 = load_b_f32(w_qkv, D_, n0 + 16, 0, lane);
    v16h b2 = load_b_f32(w_qkv, D_, n0 + 32, 0, lane);
    v16h b3 = load_b_f32(w_qkv, D_, n0 + 48, 0, lane);

    for (int k0 = 32; k0 < D_; k0 += 32) {
        // issue next-iteration loads first (latency hidden behind 8 WMMAs)
        v16h na0 = load_a_f32(x, D_, m0, k0, lane);
        v16h na1 = load_a_f32(x, D_, m0 + 16, k0, lane);
        v16h nb0 = load_b_f32(w_qkv, D_, n0 + 0,  k0, lane);
        v16h nb1 = load_b_f32(w_qkv, D_, n0 + 16, k0, lane);
        v16h nb2 = load_b_f32(w_qkv, D_, n0 + 32, k0, lane);
        v16h nb3 = load_b_f32(w_qkv, D_, n0 + 48, k0, lane);
        acc[0][0] = wmma32(a0, b0, acc[0][0]);
        acc[0][1] = wmma32(a0, b1, acc[0][1]);
        acc[0][2] = wmma32(a0, b2, acc[0][2]);
        acc[0][3] = wmma32(a0, b3, acc[0][3]);
        acc[1][0] = wmma32(a1, b0, acc[1][0]);
        acc[1][1] = wmma32(a1, b1, acc[1][1]);
        acc[1][2] = wmma32(a1, b2, acc[1][2]);
        acc[1][3] = wmma32(a1, b3, acc[1][3]);
        a0 = na0; a1 = na1; b0 = nb0; b1 = nb1; b2 = nb2; b3 = nb3;
    }
    // tail k-step
    acc[0][0] = wmma32(a0, b0, acc[0][0]);
    acc[0][1] = wmma32(a0, b1, acc[0][1]);
    acc[0][2] = wmma32(a0, b2, acc[0][2]);
    acc[0][3] = wmma32(a0, b3, acc[0][3]);
    acc[1][0] = wmma32(a1, b0, acc[1][0]);
    acc[1][1] = wmma32(a1, b1, acc[1][1]);
    acc[1][2] = wmma32(a1, b2, acc[1][2]);
    acc[1][3] = wmma32(a1, b3, acc[1][3]);

    const int col = lane & 15;
    const int hi8 = (lane >> 4) * 8;
#pragma unroll
    for (int i = 0; i < 2; ++i) {
#pragma unroll
        for (int t = 0; t < 4; ++t) {
            const int n = n0 + t * 16 + col;
            const int which = n >> 10;      // 0=q 1=k 2=v
            const int hh = (n >> 6) & 15;   // head
            const int d = n & 63;           // dim in head
#pragma unroll
            for (int r = 0; r < 8; ++r) {
                const int m   = m0 + i * 16 + r + hi8;  // flat token in [0, B*N)
                const int bi  = m >> 11;
                const int tok = m & (N_ - 1);
                const int bh  = bi * H_ + hh;
                const _Float16 h = (_Float16)acc[i][t][r];
                if (which == 0)      qb[((long)(bh * N_ + tok)) * HD_ + d] = h;
                else if (which == 1) kb[((long)(bh * N_ + tok)) * HD_ + d] = h;
                else                 vT[((long)(bh * HD_ + d)) * N_ + tok] = h;
            }
        }
    }
}

// ---- kernel 2: flash attention ---------------------------------------------
// grid (B*H, N/16/4), block (32,4): each wave owns a 16-row query block and
// streams all 2048 keys in blocks of 32 with online softmax.
__global__ __launch_bounds__(128) void attn_kernel(const int* __restrict__ mask,
                                                   const _Float16* __restrict__ qb,
                                                   const _Float16* __restrict__ kb,
                                                   const _Float16* __restrict__ vT,
                                                   _Float16* __restrict__ ob) {
    __shared__ _Float16 lds_p[4][16 * 32];   // per-wave P tile (1KB each)
    const int lane = threadIdx.x;
    const int wave = threadIdx.y;
    const int bh = blockIdx.x;                       // 0..31
    const int q0 = (blockIdx.y * 4 + wave) * 16;     // query row base
    const int b  = bh >> 4;
    const int h  = bh & 15;
    const float scale = 0.125f;                      // 1/sqrt(64)

    const _Float16* qbase = qb + (long)bh * N_ * HD_;
    const _Float16* kbase = kb + (long)bh * N_ * HD_;
    const _Float16* vbase = vT + (long)bh * HD_ * N_;

    const int col = lane & 15;
    const int hi8 = (lane >> 4) * 8;

    // hoisted mask row pointers (one per accumulator row this lane owns)
    const int* mrowp[8];
#pragma unroll
    for (int r = 0; r < 8; ++r)
        mrowp[r] = mask + (long)b * N_ * N_ + (long)(q0 + r + hi8) * N_ + col;

    // Q fragments for the whole kernel: d 0..31 and 32..63.
    const v16h aQ0 = load_a_f16(qbase, HD_, q0, 0, lane);
    const v16h aQ1 = load_a_f16(qbase, HD_, q0, 32, lane);

    v8f o0 = {}, o1 = {}, o2 = {}, o3 = {};
    float mrow[8], lrow[8];
#pragma unroll
    for (int r = 0; r < 8; ++r) { mrow[r] = -3.0e38f; lrow[r] = 0.0f; }

    _Float16* P = lds_p[wave];

    for (int j0 = 0; j0 < N_; j0 += 32) {
        // 1) issue mask loads for this block first (HBM latency hides under WMMA),
        //    and prefetch next block's mask line.
        int mv0[8], mv1[8];
#pragma unroll
        for (int r = 0; r < 8; ++r) {
            const int* mp = mrowp[r] + j0;
            mv0[r] = mp[0];
            mv1[r] = mp[16];
            __builtin_prefetch(mp + 32, 0, 1);       // global_prefetch_b8
        }

        // 2) S = Q @ K^T for 32 keys: two 16x16 f32 tiles.
        v8f s0 = {}, s1 = {};
        { v16h bk = load_b_f16(kbase, HD_, j0,      0,  lane); s0 = wmma32(aQ0, bk, s0); }
        { v16h bk = load_b_f16(kbase, HD_, j0,      32, lane); s0 = wmma32(aQ1, bk, s0); }
        { v16h bk = load_b_f16(kbase, HD_, j0 + 16, 0,  lane); s1 = wmma32(aQ0, bk, s1); }
        { v16h bk = load_b_f16(kbase, HD_, j0 + 16, 32, lane); s1 = wmma32(aQ1, bk, s1); }

        // 3) scale + mask + online softmax, per row held by this lane.
#pragma unroll
        for (int r = 0; r < 8; ++r) {
            float x0 = (mv0[r] == 0) ? -10000.0f : s0[r] * scale;
            float x1 = (mv1[r] == 0) ? -10000.0f : s1[r] * scale;
            float mx = fmaxf(x0, x1);
            mx = fmaxf(mx, __shfl_xor(mx, 1));
            mx = fmaxf(mx, __shfl_xor(mx, 2));
            mx = fmaxf(mx, __shfl_xor(mx, 4));
            mx = fmaxf(mx, __shfl_xor(mx, 8));   // row-max over 16 cols (stays in half)
            const float mnew  = fmaxf(mrow[r], mx);
            const float alpha = __expf(mrow[r] - mnew);
            const float p0 = __expf(x0 - mnew);
            const float p1 = __expf(x1 - mnew);
            float rs = p0 + p1;
            rs += __shfl_xor(rs, 1);
            rs += __shfl_xor(rs, 2);
            rs += __shfl_xor(rs, 4);
            rs += __shfl_xor(rs, 8);
            lrow[r] = lrow[r] * alpha + rs;
            mrow[r] = mnew;
            o0[r] *= alpha; o1[r] *= alpha; o2[r] *= alpha; o3[r] *= alpha;
            const int prow = r + hi8;
            P[prow * 32 + col]      = (_Float16)p0;
            P[prow * 32 + 16 + col] = (_Float16)p1;
        }

        // 4) C-layout -> A-layout transpose of P through this wave's LDS tile.
        asm volatile("s_wait_dscnt 0" ::: "memory");
        const v16h aP = load_a_lds(P, lane);

        // 5) O += P @ V  (V transposed: column d contiguous over keys).
        { v16h bv = load_b_f16(vbase, N_, 0,  j0, lane); o0 = wmma32(aP, bv, o0); }
        { v16h bv = load_b_f16(vbase, N_, 16, j0, lane); o1 = wmma32(aP, bv, o1); }
        { v16h bv = load_b_f16(vbase, N_, 32, j0, lane); o2 = wmma32(aP, bv, o2); }
        { v16h bv = load_b_f16(vbase, N_, 48, j0, lane); o3 = wmma32(aP, bv, o3); }
        asm volatile("s_wait_dscnt 0" ::: "memory");  // P reads done before next overwrite
    }

    // epilogue: divide by row sums, store f16 token-major (B,N,D) for proj GEMM.
#pragma unroll
    for (int r = 0; r < 8; ++r) {
        const float inv = 1.0f / lrow[r];
        const int tok = q0 + r + hi8;
        const long base = ((long)(b * N_ + tok)) * D_ + h * HD_;
        ob[base + 0  + col] = (_Float16)(o0[r] * inv);
        ob[base + 16 + col] = (_Float16)(o1[r] * inv);
        ob[base + 32 + col] = (_Float16)(o2[r] * inv);
        ob[base + 48 + col] = (_Float16)(o3[r] * inv);
    }
}

// ---- kernel 3: out = O @ w_out.T + b_out (f32 output) ----------------------
// grid (M/32, D/64), block 32: 32x64 tile, double-buffered like kernel 1.
__global__ __launch_bounds__(32) void proj_kernel(const _Float16* __restrict__ ob,
                                                  const float* __restrict__ w_out,
                                                  const float* __restrict__ b_out,
                                                  float* __restrict__ out) {
    const int lane = threadIdx.x;
    const int m0 = blockIdx.x * 32;
    const int n0 = blockIdx.y * 64;
    v8f acc[2][4] = {{v8f{}, v8f{}, v8f{}, v8f{}}, {v8f{}, v8f{}, v8f{}, v8f{}}};

    v16h a0 = load_a_f16(ob, D_, m0, 0, lane);
    v16h a1 = load_a_f16(ob, D_, m0 + 16, 0, lane);
    v16h b0 = load_b_f32(w_out, D_, n0 + 0,  0, lane);
    v16h b1 = load_b_f32(w_out, D_, n0 + 16, 0, lane);
    v16h b2 = load_b_f32(w_out, D_, n0 + 32, 0, lane);
    v16h b3 = load_b_f32(w_out, D_, n0 + 48, 0, lane);

    for (int k0 = 32; k0 < D_; k0 += 32) {
        v16h na0 = load_a_f16(ob, D_, m0, k0, lane);
        v16h na1 = load_a_f16(ob, D_, m0 + 16, k0, lane);
        v16h nb0 = load_b_f32(w_out, D_, n0 + 0,  k0, lane);
        v16h nb1 = load_b_f32(w_out, D_, n0 + 16, k0, lane);
        v16h nb2 = load_b_f32(w_out, D_, n0 + 32, k0, lane);
        v16h nb3 = load_b_f32(w_out, D_, n0 + 48, k0, lane);
        acc[0][0] = wmma32(a0, b0, acc[0][0]);
        acc[0][1] = wmma32(a0, b1, acc[0][1]);
        acc[0][2] = wmma32(a0, b2, acc[0][2]);
        acc[0][3] = wmma32(a0, b3, acc[0][3]);
        acc[1][0] = wmma32(a1, b0, acc[1][0]);
        acc[1][1] = wmma32(a1, b1, acc[1][1]);
        acc[1][2] = wmma32(a1, b2, acc[1][2]);
        acc[1][3] = wmma32(a1, b3, acc[1][3]);
        a0 = na0; a1 = na1; b0 = nb0; b1 = nb1; b2 = nb2; b3 = nb3;
    }
    acc[0][0] = wmma32(a0, b0, acc[0][0]);
    acc[0][1] = wmma32(a0, b1, acc[0][1]);
    acc[0][2] = wmma32(a0, b2, acc[0][2]);
    acc[0][3] = wmma32(a0, b3, acc[0][3]);
    acc[1][0] = wmma32(a1, b0, acc[1][0]);
    acc[1][1] = wmma32(a1, b1, acc[1][1]);
    acc[1][2] = wmma32(a1, b2, acc[1][2]);
    acc[1][3] = wmma32(a1, b3, acc[1][3]);

    const int col = lane & 15;
    const int hi8 = (lane >> 4) * 8;
#pragma unroll
    for (int i = 0; i < 2; ++i) {
#pragma unroll
        for (int t = 0; t < 4; ++t) {
            const int n = n0 + t * 16 + col;
            const float bias = b_out[n];
#pragma unroll
            for (int r = 0; r < 8; ++r) {
                const int m = m0 + i * 16 + r + hi8;
                out[(long)m * D_ + n] = acc[i][t][r] + bias;
            }
        }
    }
}

extern "C" void kernel_launch(void* const* d_in, const int* in_sizes, int n_in,
                              void* d_out, int out_size, void* d_ws, size_t ws_size,
                              hipStream_t stream) {
    (void)in_sizes; (void)n_in; (void)out_size; (void)ws_size;
    const float* x     = (const float*)d_in[0];
    const int*   mask  = (const int*)d_in[1];
    const float* w_qkv = (const float*)d_in[2];
    const float* w_out = (const float*)d_in[3];
    const float* b_out = (const float*)d_in[4];
    float* out = (float*)d_out;

    // workspace: 4 x 8MB f16 buffers (Q, K, V^T, O) = 32MB
    const size_t SEG = (size_t)B_ * H_ * N_ * HD_;   // 4M halves
    _Float16* qbuf  = (_Float16*)d_ws;
    _Float16* kbuf  = qbuf + SEG;
    _Float16* vTbuf = kbuf + SEG;
    _Float16* obuf  = vTbuf + SEG;

    const int M = B_ * N_;                            // 4096
    qkv_kernel <<<dim3(M / 32, (3 * D_) / 64), dim3(32),    0, stream>>>(x, w_qkv, qbuf, kbuf, vTbuf);
    attn_kernel<<<dim3(B_ * H_, N_ / 16 / 4),  dim3(32, 4), 0, stream>>>(mask, qbuf, kbuf, vTbuf, obuf);
    proj_kernel<<<dim3(M / 32, D_ / 64),       dim3(32),    0, stream>>>(obuf, w_out, b_out, out);
}
